// Model_31988916420722
// MI455X (gfx1250) — matrix operand order repaired and checked by reading, not compiled, not run on hardware
//
#include <hip/hip_runtime.h>
#include <hip/hip_bf16.h>

typedef __attribute__((ext_vector_type(16))) _Float16 v16h;
typedef __attribute__((ext_vector_type(8)))  float    v8f;

#define T_STRIDE 33   // padded LDS row stride (floats) to dodge bank conflicts

__device__ __forceinline__ float fast_rcp(float x) {
    return __builtin_amdgcn_rcpf(x);        // v_rcp_f32, no IEEE div expansion
}
__device__ __forceinline__ float sigmoidf_(float x) {
    return fast_rcp(1.0f + __expf(-x));     // v_exp_f32 + v_rcp_f32
}
__device__ __forceinline__ float tanhf_(float x) {
    // 2*sigmoid(2x)-1 : saturates correctly at both ends, no inf/inf
    return fmaf(2.0f, sigmoidf_(2.0f * x), -1.0f);
}
__device__ __forceinline__ void atomic_add_f32(float* addr, float v) {
    unsafeAtomicAdd(addr, v);   // global_atomic_add_f32
}

// ---------------- graph preprocessing ----------------

__global__ void k_deg_init(float* deg, int N) {
    int i = blockIdx.x * blockDim.x + threadIdx.x;
    if (i < N) deg[i] = 1.0f;                 // self loop
}

__global__ void k_deg_acc(const int* __restrict__ ei, float* deg, int E) {
    int e = blockIdx.x * blockDim.x + threadIdx.x;
    if (e < E) atomic_add_f32(&deg[ei[E + e]], 1.0f);   // col = target
}

__global__ void k_rsqrt(float* deg, int N) {
    int i = blockIdx.x * blockDim.x + threadIdx.x;
    if (i < N) deg[i] = rsqrtf(deg[i]);
}

// agg[n, 0:24] = dinv[n]^2 * x[n, 0:24]   (self-loop term)
__global__ void k_agg_init(const float* __restrict__ x, const float* __restrict__ dinv,
                           float* __restrict__ agg, int N) {
    int idx = blockIdx.x * blockDim.x + threadIdx.x;
    if (idx < N * 24) {
        int n = idx / 24;
        float d = dinv[n];
        agg[idx] = d * d * x[idx];
    }
}

// agg[col, :] += dinv[row]*dinv[col] * x[row, :]   (24 features, F_IN*T)
__global__ void k_agg_edges(const float* __restrict__ x, const int* __restrict__ ei,
                            const float* __restrict__ dinv, float* __restrict__ agg, int E) {
    int e = blockIdx.x * blockDim.x + threadIdx.x;
    if (e >= E) return;
    int r = ei[e];
    int c = ei[E + e];
    float w = dinv[r] * dinv[c];
    const float4* xr = reinterpret_cast<const float4*>(x + (size_t)r * 24);
    float* ac = agg + (size_t)c * 24;
#pragma unroll
    for (int q = 0; q < 6; ++q) {
        float4 xv = xr[q];
        atomic_add_f32(ac + q * 4 + 0, w * xv.x);
        atomic_add_f32(ac + q * 4 + 1, w * xv.y);
        atomic_add_f32(ac + q * 4 + 2, w * xv.z);
        atomic_add_f32(ac + q * 4 + 3, w * xv.w);
    }
}

// ---------------- recurrent GRU with WMMA ----------------
//
// Fold: [gcn_g | H] @ Lg + lbg  =  agg2 @ (Wg @ Lg[0:32]) + H @ Lg[32:64] + (bg @ Lg[0:32] + lbg)
// Per gate & per 16-col output tile, each lane keeps its column of the folded
// 2x32 matrix (m0,m1), folded bias c, and the loop-invariant B fragment of Lg[32:64].

__device__ __forceinline__ void gate_setup(
    const float* __restrict__ W, const float* __restrict__ b,
    const float* __restrict__ L, const float* __restrict__ lb,
    int col0, int hi,
    float& m00, float& m01, float& c0,
    float& m10, float& m11, float& c1,
    v16h& B0, v16h& B1)
{
    m00 = 0.f; m01 = 0.f; c0 = 0.f;
    m10 = 0.f; m11 = 0.f; c1 = 0.f;
    const int n0 = col0, n1 = col0 + 16;
#pragma unroll
    for (int k = 0; k < 32; ++k) {
        float l0 = L[k * 32 + n0], l1 = L[k * 32 + n1];
        float w0 = W[k], w1 = W[32 + k], bk = b[k];
        m00 = fmaf(w0, l0, m00); m01 = fmaf(w1, l0, m01); c0 = fmaf(bk, l0, c0);
        m10 = fmaf(w0, l1, m10); m11 = fmaf(w1, l1, m11); c1 = fmaf(bk, l1, c1);
    }
    c0 += lb[n0]; c1 += lb[n1];
    // B fragment (32x16 f16): lanes 0-15 col N=lane K=0..15, lanes 16-31 K=16..31
#pragma unroll
    for (int h = 0; h < 16; ++h) {
        int K = hi * 16 + h;
        B0[h] = (_Float16)L[(32 + K) * 32 + n0];
        B1[h] = (_Float16)L[(32 + K) * 32 + n1];
    }
}

// pre-activation = WMMA(A, B, C) where C[v] = a0[v]*M0 + a1[v]*M1 + Cb  (folded agg2 term)
#define PREACT(Afrag, M0_, M1_, Cb_, Bfrag, DST) do {                              \
    v8f cacc;                                                                      \
    _Pragma("unroll")                                                              \
    for (int v = 0; v < 8; ++v)                                                    \
        cacc[v] = fmaf(a0[v], (M0_), fmaf(a1[v], (M1_), (Cb_)));                   \
    DST = __builtin_amdgcn_wmma_f32_16x16x32_f16(                                  \
        false, (Afrag), false, (Bfrag), (short)0, cacc, false, false);             \
} while (0)

__global__ __launch_bounds__(32)
void a3t_recurrent(const float* __restrict__ agg,
                   const float* __restrict__ Wz, const float* __restrict__ bz,
                   const float* __restrict__ Lz, const float* __restrict__ lbz,
                   const float* __restrict__ Wr, const float* __restrict__ br,
                   const float* __restrict__ Lr, const float* __restrict__ lbr,
                   const float* __restrict__ Wh, const float* __restrict__ bh,
                   const float* __restrict__ Lh, const float* __restrict__ lbh,
                   const float* __restrict__ att,
                   const float* __restrict__ Wout, const float* __restrict__ bout,
                   float* __restrict__ out, int N)
{
    __shared__ float s_agg[16 * 24];        // aggregated features for this 16-node tile
    __shared__ float s_t[16 * T_STRIDE];    // transpose staging [M][K]
    __shared__ float s_p[12];               // softmax(att)

    const int lane  = threadIdx.x;
    const int hi    = lane >> 4;            // half-wave select
    const int col0  = lane & 15;
    const int nbase = blockIdx.x * 16;

    for (int i = lane; i < 16 * 24; i += 32) {
        int g = nbase * 24 + i;
        s_agg[i] = (g < N * 24) ? agg[g] : 0.0f;
    }
    if (lane < 12) {
        float mx = att[0];
#pragma unroll
        for (int t = 1; t < 12; ++t) mx = fmaxf(mx, att[t]);
        float s = 0.f;
#pragma unroll
        for (int t = 0; t < 12; ++t) s += __expf(att[t] - mx);
        s_p[lane] = __expf(att[lane] - mx) * fast_rcp(s);
    }

    float mz00, mz01, cz0, mz10, mz11, cz1; v16h Bz0, Bz1;
    float mr00, mr01, cr0, mr10, mr11, cr1; v16h Br0, Br1;
    float mh00, mh01, ch0, mh10, mh11, ch1; v16h Bh0, Bh1;
    gate_setup(Wz, bz, Lz, lbz, col0, hi, mz00, mz01, cz0, mz10, mz11, cz1, Bz0, Bz1);
    gate_setup(Wr, br, Lr, lbr, col0, hi, mr00, mr01, cr0, mr10, mr11, cr1, Br0, Br1);
    gate_setup(Wh, bh, Lh, lbh, col0, hi, mh00, mh01, ch0, mh10, mh11, ch1, Bh0, Bh1);

    __syncthreads();

    // H state & attention accumulator in C/D layout: VGPR v -> row M=v+8*hi, col = col0 (+16)
    v8f Hlo = {}, Hhi = {}, Alo = {}, Ahi = {};
    v16h hA = {};   // H in A-fragment layout (f16); H0 = 0

#pragma unroll 1
    for (int t = 0; t < 12; ++t) {
        float a0[8], a1[8];
#pragma unroll
        for (int v = 0; v < 8; ++v) {
            int m = v + 8 * hi;
            a0[v] = s_agg[m * 24 + t];        // x[n,0,t] aggregated
            a1[v] = s_agg[m * 24 + 12 + t];   // x[n,1,t] aggregated
        }

        v8f Zlo, Zhi, Rlo, Rhi, Tlo, Thi;
        PREACT(hA, mz00, mz01, cz0, Bz0, Zlo);
        PREACT(hA, mz10, mz11, cz1, Bz1, Zhi);
        PREACT(hA, mr00, mr01, cr0, Br0, Rlo);
        PREACT(hA, mr10, mr11, cr1, Br1, Rhi);
#pragma unroll
        for (int v = 0; v < 8; ++v) {
            Zlo[v] = sigmoidf_(Zlo[v]); Zhi[v] = sigmoidf_(Zhi[v]);
            Rlo[v] = sigmoidf_(Rlo[v]); Rhi[v] = sigmoidf_(Rhi[v]);
        }

        // H*R : C/D layout -> A-fragment via LDS transpose
        __syncthreads();
#pragma unroll
        for (int v = 0; v < 8; ++v) {
            int m = v + 8 * hi;
            s_t[m * T_STRIDE + col0]      = Hlo[v] * Rlo[v];
            s_t[m * T_STRIDE + col0 + 16] = Hhi[v] * Rhi[v];
        }
        __syncthreads();
        v16h hrA;
#pragma unroll
        for (int h = 0; h < 16; ++h) {
            int K = hi * 16 + h;
            hrA[h] = (_Float16)s_t[col0 * T_STRIDE + K];
        }

        PREACT(hrA, mh00, mh01, ch0, Bh0, Tlo);
        PREACT(hrA, mh10, mh11, ch1, Bh1, Thi);

        float pt = s_p[t];
#pragma unroll
        for (int v = 0; v < 8; ++v) {
            float tl = tanhf_(Tlo[v]);
            float th = tanhf_(Thi[v]);
            Hlo[v] = Zlo[v] * Hlo[v] + (1.0f - Zlo[v]) * tl;
            Hhi[v] = Zhi[v] * Hhi[v] + (1.0f - Zhi[v]) * th;
            Alo[v] = fmaf(pt, Hlo[v], Alo[v]);
            Ahi[v] = fmaf(pt, Hhi[v], Ahi[v]);
        }

        // new H -> A-fragment for next step
        __syncthreads();
#pragma unroll
        for (int v = 0; v < 8; ++v) {
            int m = v + 8 * hi;
            s_t[m * T_STRIDE + col0]      = Hlo[v];
            s_t[m * T_STRIDE + col0 + 16] = Hhi[v];
        }
        __syncthreads();
#pragma unroll
        for (int h = 0; h < 16; ++h) {
            int K = hi * 16 + h;
            hA[h] = (_Float16)s_t[col0 * T_STRIDE + K];
        }
    }

    // epilogue: out = relu(Hacc) @ W_out + b_out
    __syncthreads();
#pragma unroll
    for (int v = 0; v < 8; ++v) {
        int m = v + 8 * hi;
        s_t[m * T_STRIDE + col0]      = fmaxf(Alo[v], 0.0f);
        s_t[m * T_STRIDE + col0 + 16] = fmaxf(Ahi[v], 0.0f);
    }
    __syncthreads();
    for (int idx = lane; idx < 16 * 12; idx += 32) {
        int m = idx / 12, t = idx - m * 12;
        int node = nbase + m;
        if (node < N) {
            float acc = bout[t];
#pragma unroll
            for (int n = 0; n < 32; ++n)
                acc = fmaf(s_t[m * T_STRIDE + n], Wout[n * 12 + t], acc);
            out[node * 12 + t] = acc;
        }
    }
}

// ---------------- launch ----------------

extern "C" void kernel_launch(void* const* d_in, const int* in_sizes, int n_in,
                              void* d_out, int out_size, void* d_ws, size_t ws_size,
                              hipStream_t stream)
{
    const float* x    = (const float*)d_in[0];
    const int*   ei   = (const int*)  d_in[1];
    const float* Wz   = (const float*)d_in[2];
    const float* bz   = (const float*)d_in[3];
    const float* Lz   = (const float*)d_in[4];
    const float* lbz  = (const float*)d_in[5];
    const float* Wr   = (const float*)d_in[6];
    const float* br   = (const float*)d_in[7];
    const float* Lr   = (const float*)d_in[8];
    const float* lbr  = (const float*)d_in[9];
    const float* Wh   = (const float*)d_in[10];
    const float* bh   = (const float*)d_in[11];
    const float* Lh   = (const float*)d_in[12];
    const float* lbh  = (const float*)d_in[13];
    const float* att  = (const float*)d_in[14];
    const float* Wout = (const float*)d_in[15];
    const float* bout = (const float*)d_in[16];
    float* out = (float*)d_out;

    const int N = in_sizes[0] / 24;   // x is (N, F_IN=2, T=12)
    const int E = in_sizes[1] / 2;    // edge_index is (2, E)

    float* dinv = (float*)d_ws;                          // N floats (degree, then rsqrt)
    size_t aggOff = (((size_t)N) + 127) & ~(size_t)127;
    float* agg = dinv + aggOff;                          // N*24 floats

    dim3 b256(256);
    k_deg_init <<<dim3((N + 255) / 256),        b256, 0, stream>>>(dinv, N);
    k_deg_acc  <<<dim3((E + 255) / 256),        b256, 0, stream>>>(ei, dinv, E);
    k_rsqrt    <<<dim3((N + 255) / 256),        b256, 0, stream>>>(dinv, N);
    k_agg_init <<<dim3((N * 24 + 255) / 256),   b256, 0, stream>>>(x, dinv, agg, N);
    k_agg_edges<<<dim3((E + 255) / 256),        b256, 0, stream>>>(x, ei, dinv, agg, E);

    a3t_recurrent<<<dim3((N + 15) / 16), dim3(32), 0, stream>>>(
        agg, Wz, bz, Lz, lbz, Wr, br, Lr, lbr, Wh, bh, Lh, lbh,
        att, Wout, bout, out, N);
}
